// CausalCrossConditionalSelfAttention_75041668596239
// MI455X (gfx1250) — matrix operand order repaired and checked by reading, not compiled, not run on hardware
//
#include <hip/hip_runtime.h>
#include <hip/hip_bf16.h>

typedef __attribute__((ext_vector_type(16))) _Float16 v16h;
typedef __attribute__((ext_vector_type(8)))  _Float16 v8h;
typedef __attribute__((ext_vector_type(8)))  float    v8f;

#define EMBED 1024
#define NHEAD 16
#define HDIM  64
#define BATCH 2
#define SEQ   2048

// ---------------------------------------------------------------- fp32 -> f16
__global__ void cvt_f32_to_f16(const float* __restrict__ src,
                               _Float16* __restrict__ dst, int n) {
    int i = blockIdx.x * blockDim.x + threadIdx.x;
    if (i < n) dst[i] = (_Float16)src[i];
}

// ---------------------------------------------------------------- GEMM y = X @ W^T + bias
// 32x32 output tile per wave: 2 A-frags x 2 B-frags -> 4 WMMAs per K-step
// (doubles FLOP/byte vs 16x16 tiling; loop stays load->wmma pipelined).
// X: [M,K] f16 row-major. W: [N,K] f16 row-major (Linear weight, y = x W^T).
// out_mode 0: f16 scatter to dst16 as [B, NHEAD, T, HDIM]      (Q, K)
// out_mode 1: f16 scatter to dst16 as [B, NHEAD, HDIM, T]      (V transposed)
// out_mode 2: f32 dense [M,N] to dst32                         (final projection)
__global__ __launch_bounds__(128)
void gemm_xwt(const _Float16* __restrict__ X, const _Float16* __restrict__ W,
              const float* __restrict__ bias, _Float16* __restrict__ dst16,
              float* __restrict__ dst32, int M, int N, int K, int out_mode)
{
    const int lane = threadIdx.x & 31;
    const int wave = threadIdx.x >> 5;
    const int tile = blockIdx.x * 4 + wave;
    const int ntn  = N >> 5;                 // tiles along N (32 wide)
    const int tm   = tile / ntn;
    const int tn   = tile % ntn;
    const int h    = lane >> 4;              // K-subgroup select
    const int l    = lane & 15;              // A: row m, B: col n

    const _Float16* arow0 = X + (size_t)(tm * 32 + l) * K;
    const _Float16* arow1 = arow0 + (size_t)16 * K;
    const _Float16* brow0 = W + (size_t)(tn * 32 + l) * K;
    const _Float16* brow1 = brow0 + (size_t)16 * K;

    v8f acc[2][2] = {};
    for (int k0 = 0; k0 < K; k0 += 32) {
        union { v16h v; v8h p[2]; } a0, a1, b0, b1;
        a0.p[0] = *(const v8h*)(arow0 + k0 + 8 * h);
        a0.p[1] = *(const v8h*)(arow0 + k0 + 16 + 8 * h);
        a1.p[0] = *(const v8h*)(arow1 + k0 + 8 * h);
        a1.p[1] = *(const v8h*)(arow1 + k0 + 16 + 8 * h);
        b0.p[0] = *(const v8h*)(brow0 + k0 + 8 * h);
        b0.p[1] = *(const v8h*)(brow0 + k0 + 16 + 8 * h);
        b1.p[0] = *(const v8h*)(brow1 + k0 + 8 * h);
        b1.p[1] = *(const v8h*)(brow1 + k0 + 16 + 8 * h);
        acc[0][0] = __builtin_amdgcn_wmma_f32_16x16x32_f16(
            false, a0.v, false, b0.v, (short)0, acc[0][0], false, false);
        acc[0][1] = __builtin_amdgcn_wmma_f32_16x16x32_f16(
            false, a0.v, false, b1.v, (short)0, acc[0][1], false, false);
        acc[1][0] = __builtin_amdgcn_wmma_f32_16x16x32_f16(
            false, a1.v, false, b0.v, (short)0, acc[1][0], false, false);
        acc[1][1] = __builtin_amdgcn_wmma_f32_16x16x32_f16(
            false, a1.v, false, b1.v, (short)0, acc[1][1], false, false);
    }

#pragma unroll
    for (int sn = 0; sn < 2; ++sn) {
        const int col = tn * 32 + sn * 16 + l;
        const float bv = bias ? bias[col] : 0.0f;
#pragma unroll
        for (int sm = 0; sm < 2; ++sm) {
#pragma unroll
            for (int r = 0; r < 8; ++r) {
                int row   = tm * 32 + sm * 16 + 8 * h + r;   // token index over B*T
                float val = acc[sm][sn][r] + bv;
                if (out_mode == 2) {
                    dst32[(size_t)row * N + col] = val;
                } else {
                    int bb = row / SEQ, t = row % SEQ;
                    int hd = col / HDIM, d = col % HDIM;
                    if (out_mode == 0)
                        dst16[(((size_t)bb * NHEAD + hd) * SEQ + t) * HDIM + d] = (_Float16)val;
                    else
                        dst16[(((size_t)bb * NHEAD + hd) * HDIM + d) * SEQ + t] = (_Float16)val;
                }
            }
        }
    }
}

// ---------------------------------------------------------------- flash attention
// One wave handles one (b, h, 16-query tile); online softmax over 32-key chunks.
__global__ __launch_bounds__(128)
void attn_kernel(const _Float16* __restrict__ qh, const _Float16* __restrict__ kh,
                 const _Float16* __restrict__ vt, const unsigned char* __restrict__ pad,
                 _Float16* __restrict__ yh)
{
    __shared__ __align__(16) _Float16 plds[4][16][40];   // per-wave P staging (padded rows)

    const int lane  = threadIdx.x & 31;
    const int wave  = threadIdx.x >> 5;
    const int wid   = blockIdx.x * 4 + wave;     // 0 .. B*NHEAD*(SEQ/16)-1
    const int qtile = wid & 127;                 // SEQ/16 = 128
    const int bh    = wid >> 7;                  // b*NHEAD + head
    const int b     = bh >> 4;
    const int q0    = qtile * 16;
    const int h     = lane >> 4;
    const int l     = lane & 15;

    const _Float16* Qbase = qh + ((size_t)bh * SEQ + q0) * HDIM;
    const _Float16* Kbase = kh + (size_t)bh * SEQ * HDIM;
    const _Float16* Vbase = vt + (size_t)bh * HDIM * SEQ;

    // Q A-fragments for D-steps k0=0 and k0=32 (loaded once)
    union { v16h v; v8h p[2]; } qa0, qa1;
    {
        const _Float16* qrow = Qbase + (size_t)l * HDIM;
        qa0.p[0] = *(const v8h*)(qrow + 8 * h);
        qa0.p[1] = *(const v8h*)(qrow + 16 + 8 * h);
        qa1.p[0] = *(const v8h*)(qrow + 32 + 8 * h);
        qa1.p[1] = *(const v8h*)(qrow + 48 + 8 * h);
    }

    float m_i[8], l_i[8];
    v8f acc[4] = {};                              // four 16-col D subtiles
#pragma unroll
    for (int r = 0; r < 8; ++r) { m_i[r] = -1e30f; l_i[r] = 0.0f; }

    const float scale = 0.125f;                   // 1/sqrt(64)
    const int kend = q0 + 16;                     // causal: keys < q0+16
    for (int kc = 0; kc < kend; kc += 32) {
        // ---- S = Q K^T for key chunk [kc, kc+32) : two 16x16 f32 tiles ----
        v8f s[2] = {};
#pragma unroll
        for (int sub = 0; sub < 2; ++sub) {
            const _Float16* krow = Kbase + (size_t)(kc + sub * 16 + l) * HDIM;
            union { v16h v; v8h p[2]; } kb0, kb1;
            kb0.p[0] = *(const v8h*)(krow + 8 * h);
            kb0.p[1] = *(const v8h*)(krow + 16 + 8 * h);
            kb1.p[0] = *(const v8h*)(krow + 32 + 8 * h);
            kb1.p[1] = *(const v8h*)(krow + 48 + 8 * h);
            s[sub] = __builtin_amdgcn_wmma_f32_16x16x32_f16(
                false, qa0.v, false, kb0.v, (short)0, s[sub], false, false);
            s[sub] = __builtin_amdgcn_wmma_f32_16x16x32_f16(
                false, qa1.v, false, kb1.v, (short)0, s[sub], false, false);
        }

        // ---- scale + causal/padding mask + row max (shfl across the 16-lane half) ----
        float rowmax[8];
#pragma unroll
        for (int r = 0; r < 8; ++r) {
            int qi = q0 + 8 * h + r;
#pragma unroll
            for (int sub = 0; sub < 2; ++sub) {
                int ki = kc + sub * 16 + l;
                float sv = s[sub][r] * scale;
                bool masked = (ki > qi) || (pad[b * SEQ + ki] != 0);
                s[sub][r] = masked ? -1e30f : sv;
            }
            float rm = fmaxf(s[0][r], s[1][r]);
            rm = fmaxf(rm, __shfl_xor(rm, 1));
            rm = fmaxf(rm, __shfl_xor(rm, 2));
            rm = fmaxf(rm, __shfl_xor(rm, 4));
            rm = fmaxf(rm, __shfl_xor(rm, 8));
            rowmax[r] = rm;
        }

        // ---- online softmax update; stage P (f16) into LDS for layout transpose ----
#pragma unroll
        for (int r = 0; r < 8; ++r) {
            float mnew  = fmaxf(m_i[r], rowmax[r]);
            float alpha = __expf(m_i[r] - mnew);
            float p0    = __expf(s[0][r] - mnew);
            float p1    = __expf(s[1][r] - mnew);
            m_i[r] = mnew;
            float rs = p0 + p1;
            rs += __shfl_xor(rs, 1);
            rs += __shfl_xor(rs, 2);
            rs += __shfl_xor(rs, 4);
            rs += __shfl_xor(rs, 8);
            l_i[r] = l_i[r] * alpha + rs;
#pragma unroll
            for (int t = 0; t < 4; ++t) acc[t][r] *= alpha;
            plds[wave][8 * h + r][l]      = (_Float16)p0;
            plds[wave][8 * h + r][16 + l] = (_Float16)p1;
        }
        asm volatile("s_wait_dscnt 0" ::: "memory");   // same-wave LDS RAW ordering

        // ---- O += P @ V  (A = P 16x32 from LDS, B = V^T contiguous in key) ----
        union { v16h v; v8h p[2]; } pa;
        pa.p[0] = *(const v8h*)(&plds[wave][l][8 * h]);
        pa.p[1] = *(const v8h*)(&plds[wave][l][16 + 8 * h]);
#pragma unroll
        for (int t = 0; t < 4; ++t) {
            const _Float16* vrow = Vbase + (size_t)(t * 16 + l) * SEQ + kc;
            union { v16h v; v8h p[2]; } vb;
            vb.p[0] = *(const v8h*)(vrow + 8 * h);
            vb.p[1] = *(const v8h*)(vrow + 16 + 8 * h);
            acc[t] = __builtin_amdgcn_wmma_f32_16x16x32_f16(
                false, pa.v, false, vb.v, (short)0, acc[t], false, false);
        }
    }

    // ---- normalize and write y (f16, [B,T,C] with C = head*64 + d) ----
    const int head = bh & 15;
#pragma unroll
    for (int r = 0; r < 8; ++r) {
        float inv = 1.0f / l_i[r];
        int trow  = q0 + 8 * h + r;
#pragma unroll
        for (int t = 0; t < 4; ++t) {
            float val = acc[t][r] * inv;
            yh[((size_t)b * SEQ + trow) * EMBED + head * HDIM + t * 16 + l] = (_Float16)val;
        }
    }
}

// ---------------------------------------------------------------- launcher
extern "C" void kernel_launch(void* const* d_in, const int* in_sizes, int n_in,
                              void* d_out, int out_size, void* d_ws, size_t ws_size,
                              hipStream_t stream) {
    (void)in_sizes; (void)n_in; (void)out_size; (void)ws_size;

    const float*         x   = (const float*)d_in[0];
    const unsigned char* pad = (const unsigned char*)d_in[1];
    const float* Wq = (const float*)d_in[2];
    const float* bq = (const float*)d_in[3];
    const float* Wk = (const float*)d_in[4];
    const float* bk = (const float*)d_in[5];
    const float* Wv = (const float*)d_in[6];
    const float* bv = (const float*)d_in[7];
    const float* Wp = (const float*)d_in[8];
    const float* bp = (const float*)d_in[9];
    float* out = (float*)d_out;

    const size_t XE = (size_t)BATCH * SEQ * EMBED;   // 4,194,304 elems
    const size_t WE = (size_t)EMBED * EMBED;         // 1,048,576 elems
    _Float16* ws  = (_Float16*)d_ws;                 // total ~48 MiB of f16
    _Float16* xh  = ws;
    _Float16* wqh = xh + XE;
    _Float16* wkh = wqh + WE;
    _Float16* wvh = wkh + WE;
    _Float16* wph = wvh + WE;
    _Float16* qh  = wph + WE;
    _Float16* kh  = qh + XE;
    _Float16* vt  = kh + XE;
    _Float16* yh  = vt + XE;

    cvt_f32_to_f16<<<(int)((XE + 255) / 256), 256, 0, stream>>>(x,  xh,  (int)XE);
    cvt_f32_to_f16<<<(int)((WE + 255) / 256), 256, 0, stream>>>(Wq, wqh, (int)WE);
    cvt_f32_to_f16<<<(int)((WE + 255) / 256), 256, 0, stream>>>(Wk, wkh, (int)WE);
    cvt_f32_to_f16<<<(int)((WE + 255) / 256), 256, 0, stream>>>(Wv, wvh, (int)WE);
    cvt_f32_to_f16<<<(int)((WE + 255) / 256), 256, 0, stream>>>(Wp, wph, (int)WE);

    const int M = BATCH * SEQ, N = EMBED, K = EMBED;
    dim3 gg((M / 32) * (N / 32) / 4), gb(128);       // 4096 tiles / 4 waves
    gemm_xwt<<<gg, gb, 0, stream>>>(xh, wqh, bq, qh, nullptr, M, N, K, 0);
    gemm_xwt<<<gg, gb, 0, stream>>>(xh, wkh, bk, kh, nullptr, M, N, K, 0);
    gemm_xwt<<<gg, gb, 0, stream>>>(xh, wvh, bv, vt, nullptr, M, N, K, 1);

    attn_kernel<<<BATCH * NHEAD * (SEQ / 16) / 4, 128, 0, stream>>>(qh, kh, vt, pad, yh);

    gemm_xwt<<<gg, gb, 0, stream>>>(yh, wph, bp, nullptr, out, M, N, K, 2);
}
